// TransformerBlock_79448305041808
// MI455X (gfx1250) — matrix operand order repaired and checked by reading, not compiled, not run on hardware
//
#include <hip/hip_runtime.h>
#include <hip/hip_bf16.h>
#include <math.h>

// ---------------------------------------------------------------------------
// Types matching CDNA5 WMMA / TDM builtin signatures
// ---------------------------------------------------------------------------
typedef _Float16 h16v __attribute__((ext_vector_type(16)));
typedef _Float16 h8v  __attribute__((ext_vector_type(8)));
typedef float    f8v  __attribute__((ext_vector_type(8)));
typedef unsigned int u32x4 __attribute__((ext_vector_type(4)));
typedef int          i32x4 __attribute__((ext_vector_type(4)));
typedef int          i32x8 __attribute__((ext_vector_type(8)));

#define DIM      1024
#define HEADS    16
#define HEAD_DIM 64
#define HIDDEN   4096
#define SEQ      2048
#define NB       4
#define CTXL     77
#define CTXD     768

#if __has_builtin(__builtin_amdgcn_tensor_load_to_lds)
#define HAVE_TDM 1
#else
#define HAVE_TDM 0
#endif

__device__ __forceinline__ f8v f8zero() {
  f8v z = {0.f, 0.f, 0.f, 0.f, 0.f, 0.f, 0.f, 0.f};
  return z;
}

// A-matrix fragment (16x32 f16, MxK): lane m (0-15) holds K = 0..15,
// lane m+16 holds K = 16..31.  Caller passes p = rowptr + (lane>>4)*16.
__device__ __forceinline__ h16v load_a_frag(const _Float16* p) {
  h8v lo = *(const h8v*)(p);
  h8v hi = *(const h8v*)(p + 8);
  h16v r;
#pragma unroll
  for (int i = 0; i < 8; ++i) { r[i] = lo[i]; r[i + 8] = hi[i]; }
  return r;
}

#if HAVE_TDM
// ---------------------------------------------------------------------------
// TDM: async DMA of a 2-D f16 tile (tile_rows x tile_cols elements) from
// global (row stride ld elements) into LDS at byte offset lds_off.
// rows_rem/cols_rem give the tensor extent from the tile origin: rows/cols
// beyond the extent are zero-filled by the TDM's OOB handling (D# spec §8.2).
// ---------------------------------------------------------------------------
__device__ __forceinline__ void tdm_load_tile(uint32_t lds_off,
                                              const _Float16* g,
                                              uint32_t rows_rem,
                                              uint32_t cols_rem,
                                              uint32_t ld,
                                              uint32_t tile_rows,
                                              uint32_t tile_cols) {
  uint64_t ga = (uint64_t)(uintptr_t)g;
  u32x4 g0 = {0u, 0u, 0u, 0u};
  g0[0] = 1u;                                       // count=1, user desc
  g0[1] = lds_off;                                  // lds_addr (bytes)
  g0[2] = (uint32_t)ga;                             // global_addr[31:0]
  g0[3] = ((uint32_t)(ga >> 32) & 0x01ffffffu) | (2u << 30);  // addr[56:32]|type=2
  i32x8 g1 = {0, 0, 0, 0, 0, 0, 0, 0};
  g1[0] = (int)(1u << 16);                          // data_size = 1 -> 2 bytes
  g1[1] = (int)((cols_rem & 0xffffu) << 16);        // tensor_dim0[15:0]
  g1[2] = (int)((cols_rem >> 16) | ((rows_rem & 0xffffu) << 16));  // dim0 hi|dim1 lo
  g1[3] = (int)((rows_rem >> 16) | (tile_cols << 16));             // dim1 hi|tile_dim0
  g1[4] = (int)tile_rows;                           // tile_dim1 (tile_dim2=0)
  g1[5] = (int)ld;                                  // tensor_dim0_stride[31:0]
  i32x4 gz = {0, 0, 0, 0};
#if defined(__clang_major__) && (__clang_major__ >= 23)
  i32x8 gz8 = {0, 0, 0, 0, 0, 0, 0, 0};
  __builtin_amdgcn_tensor_load_to_lds(g0, g1, gz, gz, gz8, 0);
#else
  __builtin_amdgcn_tensor_load_to_lds(g0, g1, gz, gz, 0);
#endif
}
#endif  // HAVE_TDM

// Fallback cooperative copy (also zero-fills OOB rows).
__device__ __forceinline__ void stage_tile_fallback(_Float16* dst,
                                                    const _Float16* g,
                                                    int rows_rem, int ld,
                                                    int tile_rows) {
  const int tid = threadIdx.x;
  const int c = (tid & 7) * 8;
  h8v zz = {(_Float16)0.f, (_Float16)0.f, (_Float16)0.f, (_Float16)0.f,
            (_Float16)0.f, (_Float16)0.f, (_Float16)0.f, (_Float16)0.f};
  for (int r = tid >> 3; r < tile_rows; r += 32) {
    h8v v = (r < rows_rem) ? *(const h8v*)(g + (size_t)r * ld + c) : zz;
    *(h8v*)(dst + r * 64 + c) = v;
  }
}

// ---------------------------------------------------------------------------
// f32 -> f16 conversion (weights, context)
// ---------------------------------------------------------------------------
__global__ __launch_bounds__(256) void cvt_f32_f16_kernel(
    const float* __restrict__ in, _Float16* __restrict__ out, int n) {
  int i = blockIdx.x * 256 + threadIdx.x;
  if (i < n) out[i] = (_Float16)in[i];
}

// ---------------------------------------------------------------------------
// LayerNorm over 1024 cols -> f16 (feeds WMMA GEMMs)
// ---------------------------------------------------------------------------
__global__ __launch_bounds__(256) void layernorm_f16_kernel(
    const float* __restrict__ x, const float* __restrict__ w,
    const float* __restrict__ b, _Float16* __restrict__ out) {
  __shared__ float red0[8];
  __shared__ float red1[8];
  __shared__ float stats[2];
  const int row = blockIdx.x;
  const int tid = threadIdx.x;
  const float* xr = x + (size_t)row * DIM;
  float4 v4 = *(const float4*)(xr + tid * 4);
  float vv[4] = {v4.x, v4.y, v4.z, v4.w};
  float s = vv[0] + vv[1] + vv[2] + vv[3];
  float s2 = vv[0] * vv[0] + vv[1] * vv[1] + vv[2] * vv[2] + vv[3] * vv[3];
#pragma unroll
  for (int m = 16; m >= 1; m >>= 1) {
    s  += __shfl_xor(s,  m, 32);
    s2 += __shfl_xor(s2, m, 32);
  }
  if ((tid & 31) == 0) { red0[tid >> 5] = s; red1[tid >> 5] = s2; }
  __syncthreads();
  if (tid == 0) {
    float a = 0.f, c = 0.f;
#pragma unroll
    for (int i = 0; i < 8; ++i) { a += red0[i]; c += red1[i]; }
    float mu = a * (1.f / DIM);
    float var = c * (1.f / DIM) - mu * mu;
    stats[0] = mu;
    stats[1] = rsqrtf(var + 1e-5f);
  }
  __syncthreads();
  const float mu = stats[0], rs = stats[1];
  _Float16* orow = out + (size_t)row * DIM;
#pragma unroll
  for (int j = 0; j < 4; ++j) {
    int c = tid * 4 + j;
    orow[c] = (_Float16)((vv[j] - mu) * rs * w[c] + b[c]);
  }
}

// ---------------------------------------------------------------------------
// WMMA GEMM: out[M,N] = A[M,K] @ W[N,K]^T + bias
//   mode 0: store f16       mode 1: exact GELU -> f16
//   mode 2: f32 + residual -> out32
// Workgroup macro-tile 64M x 128N, 8 waves, each wave a 32x32 C block
// (4 WMMA accumulators -> 2x fragment reuse).  A (64x64) and B (128x64) f16
// K-chunks are staged into LDS by the Tensor Data Mover, double-buffered:
// TDM streams chunk k+1 (TENSORcnt) while waves run 8 WMMAs on chunk k.
// ---------------------------------------------------------------------------
__global__ __launch_bounds__(256) void gemm_wmma_kernel(
    const _Float16* __restrict__ A, const _Float16* __restrict__ W,
    const float* __restrict__ bias, _Float16* __restrict__ out16,
    float* __restrict__ out32, const float* __restrict__ residual,
    int M, int N, int K, int mode) {
  __shared__ __align__(32) _Float16 sA[2][64 * 64];     // 2 x 8 KB
  __shared__ __align__(32) _Float16 sB[2][128 * 64];    // 2 x 16 KB

  const int lane = threadIdx.x & 31;
  const int wid  = threadIdx.x >> 5;
  const int wn = wid & 3;     // 32-col group within 128
  const int wm = wid >> 2;    // 32-row group within 64
  const int bm = blockIdx.x * 64;
  const int bn = blockIdx.y * 128;
  const int nk = K >> 6;      // K chunks of 64

  const _Float16* Ab = A + (size_t)bm * K;
  const _Float16* Wb = W + (size_t)bn * K;
  const uint32_t rowsA = (uint32_t)(M - bm);
  const uint32_t rowsB = (uint32_t)(N - bn);

#if HAVE_TDM
  if (wid == 0) {
    tdm_load_tile((uint32_t)(uintptr_t)&sA[0][0], Ab, rowsA, (uint32_t)K,
                  (uint32_t)K, 64, 64);
    tdm_load_tile((uint32_t)(uintptr_t)&sB[0][0], Wb, rowsB, (uint32_t)K,
                  (uint32_t)K, 128, 64);
  }
#endif

  f8v acc[2][2] = {{f8zero(), f8zero()}, {f8zero(), f8zero()}};

  int cur = 0;
  for (int kc = 0; kc < nk; ++kc) {
#if HAVE_TDM
    if (wid == 0) __builtin_amdgcn_s_wait_tensorcnt(0);
    __syncthreads();                         // chunk `cur` visible to all waves
    if (wid == 0 && kc + 1 < nk) {           // stream next chunk during compute
      const int k = (kc + 1) << 6;
      tdm_load_tile((uint32_t)(uintptr_t)&sA[cur ^ 1][0], Ab + k, rowsA,
                    (uint32_t)(K - k), (uint32_t)K, 64, 64);
      tdm_load_tile((uint32_t)(uintptr_t)&sB[cur ^ 1][0], Wb + k, rowsB,
                    (uint32_t)(K - k), (uint32_t)K, 128, 64);
    }
#else
    __syncthreads();                         // WAR on buffer 0
    stage_tile_fallback(&sA[0][0], Ab + (kc << 6), (int)rowsA, K, 64);
    stage_tile_fallback(&sB[0][0], Wb + (kc << 6), (int)rowsB, K, 128);
    __syncthreads();
#endif

    const _Float16* a_base = &sA[cur][0];
    const _Float16* b_base = &sB[cur][0];
#pragma unroll
    for (int kk = 0; kk < 64; kk += 32) {
      h16v af[2], bf[2];
#pragma unroll
      for (int ms = 0; ms < 2; ++ms) {
        const _Float16* p = a_base + (wm * 32 + ms * 16 + (lane & 15)) * 64 +
                            kk + ((lane >> 4) << 4);
        af[ms] = load_a_frag(p);
      }
#pragma unroll
      for (int ns = 0; ns < 2; ++ns) {
        const _Float16* p = b_base + (wn * 32 + ns * 16 + (lane & 15)) * 64 +
                            kk + ((lane >> 4) << 4);
        bf[ns] = *(const h16v*)p;
      }
#pragma unroll
      for (int ms = 0; ms < 2; ++ms)
#pragma unroll
        for (int ns = 0; ns < 2; ++ns)
          acc[ms][ns] = __builtin_amdgcn_wmma_f32_16x16x32_f16(
              false, af[ms], false, bf[ns], (short)0, acc[ms][ns], false, false);
    }
#if HAVE_TDM
    __syncthreads();                         // reads of `cur` done before overwrite
    cur ^= 1;
#endif
  }

  // ---- epilogue ----
  const int mhi = ((lane >> 4) << 3);
#pragma unroll
  for (int ms = 0; ms < 2; ++ms) {
#pragma unroll
    for (int ns = 0; ns < 2; ++ns) {
      const int n = bn + wn * 32 + ns * 16 + (lane & 15);
      const float bnv = bias[n];
#pragma unroll
      for (int i = 0; i < 8; ++i) {
        const int m = bm + wm * 32 + ms * 16 + mhi + i;
        if (m < M) {
          float v = acc[ms][ns][i] + bnv;
          if (mode == 1) v = 0.5f * v * (1.f + erff(v * 0.70710678118654752f));
          if (mode == 2)
            out32[(size_t)m * N + n] = v + residual[(size_t)m * N + n];
          else
            out16[(size_t)m * N + n] = (_Float16)v;
        }
      }
    }
  }
}

// ---------------------------------------------------------------------------
// Flash attention (one 16-query tile per wave, online softmax, WMMA QK^T & PV)
// Q: [B, Lq, 1024] (head hh occupies cols hh*64..), K/V: [B, Lkv, 1024]
// ---------------------------------------------------------------------------
__global__ __launch_bounds__(256, 1) void attention_wmma_kernel(
    const _Float16* __restrict__ Q, const _Float16* __restrict__ Km,
    const _Float16* __restrict__ Vm, _Float16* __restrict__ O,
    int Lq, int Lkv, int nbatch, float scale) {
  __shared__ __align__(32) _Float16 s_p[8][16][48];   // probs (16 q x 32 s)/wave
  __shared__ __align__(32) _Float16 s_vt[8][64][48];  // V^T (64 d x 32 s)/wave

  const int lane = threadIdx.x & 31;
  const int wid  = threadIdx.x >> 5;
  const int qtiles = Lq >> 4;
  const int total = nbatch * HEADS * qtiles;
  int t = blockIdx.x * 8 + wid;
  if (t >= total) t = total - 1;                  // duplicate work, identical writes
  const int qt = t % qtiles;
  const int hh = (t / qtiles) % HEADS;
  const int bb = t / (qtiles * HEADS);

  const _Float16* Qb = Q + ((size_t)(bb * Lq + qt * 16)) * DIM + hh * HEAD_DIM;
  const _Float16* Kb = Km + (size_t)bb * Lkv * DIM + hh * HEAD_DIM;
  const _Float16* Vb = Vm + (size_t)bb * Lkv * DIM + hh * HEAD_DIM;

  const _Float16* qrow = Qb + (size_t)(lane & 15) * DIM + ((lane >> 4) << 4);
  h16v q0 = load_a_frag(qrow);        // K(d) = 0..31
  h16v q1 = load_a_frag(qrow + 32);   // K(d) = 32..63

  float mrun[8], lrun[8], alpha[8];
  f8v o[4] = {f8zero(), f8zero(), f8zero(), f8zero()};
#pragma unroll
  for (int i = 0; i < 8; ++i) { mrun[i] = -1e30f; lrun[i] = 0.f; }

  const int mrow = ((lane >> 4) << 3);

  for (int s0 = 0; s0 < Lkv; s0 += 32) {
    // ---- scores: two 16x16 tiles covering keys s0..s0+31 ----
    f8v sc[2];
#pragma unroll
    for (int nt = 0; nt < 2; ++nt) {
      int srow = s0 + nt * 16 + (lane & 15);
      int src = srow < Lkv ? srow : Lkv - 1;
      const _Float16* kp = Kb + (size_t)src * DIM + ((lane >> 4) << 4);
      h16v kb0 = *(const h16v*)(kp);
      h16v kb1 = *(const h16v*)(kp + 32);
      f8v a = __builtin_amdgcn_wmma_f32_16x16x32_f16(false, q0, false, kb0,
                                                     (short)0, f8zero(), false, false);
      a = __builtin_amdgcn_wmma_f32_16x16x32_f16(false, q1, false, kb1,
                                                 (short)0, a, false, false);
      const bool oob = srow >= Lkv;
#pragma unroll
      for (int i = 0; i < 8; ++i) sc[nt][i] = oob ? -1e30f : a[i] * scale;
    }

    // ---- online softmax stats (row reduce across 16 lanes per half) ----
#pragma unroll
    for (int i = 0; i < 8; ++i) {
      float mx = fmaxf(sc[0][i], sc[1][i]);
#pragma unroll
      for (int msk = 1; msk < 16; msk <<= 1) mx = fmaxf(mx, __shfl_xor(mx, msk, 32));
      float mnew = fmaxf(mrun[i], mx);
      alpha[i] = __expf(mrun[i] - mnew);
      mrun[i] = mnew;
    }

    // ---- probabilities -> LDS (A-fragment source), row sums ----
#pragma unroll
    for (int i = 0; i < 8; ++i) {
      float p0 = __expf(sc[0][i] - mrun[i]);
      float p1 = __expf(sc[1][i] - mrun[i]);
      float rs = p0 + p1;
#pragma unroll
      for (int msk = 1; msk < 16; msk <<= 1) rs += __shfl_xor(rs, msk, 32);
      lrun[i] = lrun[i] * alpha[i] + rs;
      s_p[wid][mrow + i][(lane & 15)]      = (_Float16)p0;
      s_p[wid][mrow + i][16 + (lane & 15)] = (_Float16)p1;
    }

    // ---- stage V chunk transposed into LDS: s_vt[d][s] ----
    {
      int vrow = s0 + lane;
      int vrc = vrow < Lkv ? vrow : Lkv - 1;
      const bool vvalid = vrow < Lkv;
      const _Float16* vp = Vb + (size_t)vrc * DIM;
#pragma unroll
      for (int dd = 0; dd < 64; dd += 8) {
        h8v vv = *(const h8v*)(vp + dd);
#pragma unroll
        for (int j = 0; j < 8; ++j)
          s_vt[wid][dd + j][lane] = vvalid ? vv[j] : (_Float16)0.f;
      }
    }
    asm volatile("s_wait_dscnt 0x0" ::: "memory");   // wave-local LDS RAW

    // ---- P fragment (16x32, A layout) from LDS ----
    h16v pa;
    {
      const _Float16* pp = &s_p[wid][lane & 15][((lane >> 4) << 4)];
      h8v lo = *(const h8v*)(pp);
      h8v hi = *(const h8v*)(pp + 8);
#pragma unroll
      for (int i = 0; i < 8; ++i) { pa[i] = lo[i]; pa[i + 8] = hi[i]; }
    }

    // ---- PV: 4 output d-tiles, rescale running accumulators ----
#pragma unroll
    for (int tt = 0; tt < 4; ++tt) {
      const _Float16* vbp = &s_vt[wid][tt * 16 + (lane & 15)][((lane >> 4) << 4)];
      h16v vb = *(const h16v*)vbp;
      f8v cc;
#pragma unroll
      for (int i = 0; i < 8; ++i) cc[i] = o[tt][i] * alpha[i];
      o[tt] = __builtin_amdgcn_wmma_f32_16x16x32_f16(false, pa, false, vb,
                                                     (short)0, cc, false, false);
    }
    asm volatile("s_wait_dscnt 0x0" ::: "memory");   // LDS WAR before next chunk
  }

  // ---- normalize + store ----
  _Float16* Ob = O + ((size_t)(bb * Lq + qt * 16)) * DIM + hh * HEAD_DIM;
#pragma unroll
  for (int tt = 0; tt < 4; ++tt) {
#pragma unroll
    for (int i = 0; i < 8; ++i) {
      float val = o[tt][i] / lrun[i];
      Ob[(size_t)(mrow + i) * DIM + tt * 16 + (lane & 15)] = (_Float16)val;
    }
  }
}

// ---------------------------------------------------------------------------
// Host-side orchestration
// ---------------------------------------------------------------------------
extern "C" void kernel_launch(void* const* d_in, const int* in_sizes, int n_in,
                              void* d_out, int out_size, void* d_ws, size_t ws_size,
                              hipStream_t stream) {
  const float* x       = (const float*)d_in[0];
  const float* context = (const float*)d_in[1];
  const float* ln1_w = (const float*)d_in[2];  const float* ln1_b = (const float*)d_in[3];
  const float* ln2_w = (const float*)d_in[4];  const float* ln2_b = (const float*)d_in[5];
  const float* ln3_w = (const float*)d_in[6];  const float* ln3_b = (const float*)d_in[7];
  const float* sa_qw = (const float*)d_in[8];  const float* sa_qb = (const float*)d_in[9];
  const float* sa_kw = (const float*)d_in[10]; const float* sa_kb = (const float*)d_in[11];
  const float* sa_vw = (const float*)d_in[12]; const float* sa_vb = (const float*)d_in[13];
  const float* sa_pw = (const float*)d_in[14]; const float* sa_pb = (const float*)d_in[15];
  const float* ca_qw = (const float*)d_in[16]; const float* ca_qb = (const float*)d_in[17];
  const float* ca_kw = (const float*)d_in[18]; const float* ca_kb = (const float*)d_in[19];
  const float* ca_vw = (const float*)d_in[20]; const float* ca_vb = (const float*)d_in[21];
  const float* ca_pw = (const float*)d_in[22]; const float* ca_pb = (const float*)d_in[23];
  const float* fc1_w = (const float*)d_in[24]; const float* fc1_b = (const float*)d_in[25];
  const float* fc2_w = (const float*)d_in[26]; const float* fc2_b = (const float*)d_in[27];
  float* out = (float*)d_out;                      // running residual stream x

  const int ML = NB * SEQ;    // 8192 rows
  const int MC = NB * CTXL;   // 308 context rows

  char* ws = (char*)d_ws;
  size_t off = 0;
  auto alloc = [&](size_t bytes) -> char* {
    char* p = ws + off;
    off += (bytes + 255) & ~(size_t)255;
    return p;
  };

  // f16 weight copies
  _Float16* w_saq = (_Float16*)alloc((size_t)DIM * DIM * 2);
  _Float16* w_sak = (_Float16*)alloc((size_t)DIM * DIM * 2);
  _Float16* w_sav = (_Float16*)alloc((size_t)DIM * DIM * 2);
  _Float16* w_sap = (_Float16*)alloc((size_t)DIM * DIM * 2);
  _Float16* w_caq = (_Float16*)alloc((size_t)DIM * DIM * 2);
  _Float16* w_cak = (_Float16*)alloc((size_t)DIM * CTXD * 2);
  _Float16* w_cav = (_Float16*)alloc((size_t)DIM * CTXD * 2);
  _Float16* w_cap = (_Float16*)alloc((size_t)DIM * DIM * 2);
  _Float16* w_fc1 = (_Float16*)alloc((size_t)HIDDEN * DIM * 2);
  _Float16* w_fc2 = (_Float16*)alloc((size_t)DIM * HIDDEN * 2);
  // f16 activations
  _Float16* ctx16  = (_Float16*)alloc((size_t)MC * CTXD * 2);
  _Float16* kc16   = (_Float16*)alloc((size_t)MC * DIM * 2);
  _Float16* vc16   = (_Float16*)alloc((size_t)MC * DIM * 2);
  _Float16* h16    = (_Float16*)alloc((size_t)ML * DIM * 2);
  _Float16* q16    = (_Float16*)alloc((size_t)ML * DIM * 2);   // big region start
  _Float16* k16    = (_Float16*)alloc((size_t)ML * DIM * 2);
  _Float16* v16    = (_Float16*)alloc((size_t)ML * DIM * 2);
  _Float16* attn16 = (_Float16*)alloc((size_t)ML * DIM * 2);
  _Float16* mlp16  = q16;  // MLP hidden (8192x4096 f16 = 64MB) aliases q/k/v/attn

  auto cvt = [&](const float* src, _Float16* dst, size_t n) {
    cvt_f32_f16_kernel<<<(unsigned)((n + 255) / 256), 256, 0, stream>>>(src, dst, (int)n);
  };
  auto ln = [&](const float* src, const float* w, const float* b, _Float16* dst) {
    layernorm_f16_kernel<<<ML, 256, 0, stream>>>(src, w, b, dst);
  };
  auto gemm = [&](const _Float16* A, const _Float16* W, const float* bias,
                  _Float16* o16, float* o32, const float* res,
                  int M, int N, int K, int mode) {
    dim3 grid((M + 63) / 64, N / 128);
    gemm_wmma_kernel<<<grid, 256, 0, stream>>>(A, W, bias, o16, o32, res, M, N, K, mode);
  };
  auto attn = [&](const _Float16* Qm, const _Float16* Km, const _Float16* Vm,
                  _Float16* Om, int Lkv) {
    int tiles = NB * HEADS * (SEQ / 16);
    attention_wmma_kernel<<<(tiles + 7) / 8, 256, 0, stream>>>(
        Qm, Km, Vm, Om, SEQ, Lkv, NB, 0.125f /* 1/sqrt(64) */);
  };

  // ---- weight / context conversion ----
  cvt(sa_qw, w_saq, (size_t)DIM * DIM);   cvt(sa_kw, w_sak, (size_t)DIM * DIM);
  cvt(sa_vw, w_sav, (size_t)DIM * DIM);   cvt(sa_pw, w_sap, (size_t)DIM * DIM);
  cvt(ca_qw, w_caq, (size_t)DIM * DIM);   cvt(ca_kw, w_cak, (size_t)DIM * CTXD);
  cvt(ca_vw, w_cav, (size_t)DIM * CTXD);  cvt(ca_pw, w_cap, (size_t)DIM * DIM);
  cvt(fc1_w, w_fc1, (size_t)HIDDEN * DIM); cvt(fc2_w, w_fc2, (size_t)DIM * HIDDEN);
  cvt(context, ctx16, (size_t)MC * CTXD);

  // ---- self-attention ----
  ln(x, ln1_w, ln1_b, h16);
  gemm(h16, w_saq, sa_qb, q16, nullptr, nullptr, ML, DIM, DIM, 0);
  gemm(h16, w_sak, sa_kb, k16, nullptr, nullptr, ML, DIM, DIM, 0);
  gemm(h16, w_sav, sa_vb, v16, nullptr, nullptr, ML, DIM, DIM, 0);
  attn(q16, k16, v16, attn16, SEQ);
  gemm(attn16, w_sap, sa_pb, nullptr, out, x, ML, DIM, DIM, 2);   // out = x + proj

  // ---- cross-attention ----
  ln(out, ln2_w, ln2_b, h16);
  gemm(h16, w_caq, ca_qb, q16, nullptr, nullptr, ML, DIM, DIM, 0);
  gemm(ctx16, w_cak, ca_kb, kc16, nullptr, nullptr, MC, DIM, CTXD, 0);
  gemm(ctx16, w_cav, ca_vb, vc16, nullptr, nullptr, MC, DIM, CTXD, 0);
  attn(q16, kc16, vc16, attn16, CTXL);
  gemm(attn16, w_cap, ca_pb, nullptr, out, out, ML, DIM, DIM, 2); // out += proj

  // ---- MLP ----
  ln(out, ln3_w, ln3_b, h16);
  gemm(h16, w_fc1, fc1_b, mlp16, nullptr, nullptr, ML, HIDDEN, DIM, 1);  // GELU
  gemm(mlp16, w_fc2, fc2_b, nullptr, out, out, ML, DIM, HIDDEN, 2);      // out += fc2
}
